// HEALVAEEncoderBlock_2327872274546
// MI455X (gfx1250) — compile-verified, matrix operand order checked by reading, not run on hardware
//
#include <hip/hip_runtime.h>
#include <hip/hip_bf16.h>

#define NN 49152
#define NE 393216

typedef __bf16 bf16;
typedef __attribute__((ext_vector_type(16))) __bf16 v16bf;
typedef __attribute__((ext_vector_type(8)))  __bf16 v8bf;
typedef __attribute__((ext_vector_type(8)))  float  v8f;
typedef __attribute__((ext_vector_type(4)))  int    v4i;

// ---- CDNA5 async global->LDS staging (guarded; falls back to ds_store) -----
#if defined(__has_builtin)
#if __has_builtin(__builtin_amdgcn_global_load_async_to_lds_b128)
#define HAS_ASYNC_LDS 1
#endif
#endif
#ifndef HAS_ASYNC_LDS
#define HAS_ASYNC_LDS 0
#endif

#if HAS_ASYNC_LDS
typedef __attribute__((address_space(1))) v4i gv4i;   // global
typedef __attribute__((address_space(3))) v4i lv4i;   // LDS
#endif

__device__ __forceinline__ float gelu_f(float v) {
    return 0.5f * v * (1.0f + erff(v * 0.70710678118654752f));
}

__device__ __forceinline__ v8f wmma_bf16(v16bf a, v16bf b, v8f c) {
    // D = A(16x32 bf16) * B(32x16 bf16) + C(16x16 f32)
    return __builtin_amdgcn_wmma_f32_16x16x32_bf16(
        false, a, false, b, (short)0, c, false, false);
}

// Build a 16x32 bf16 A-fragment for one lane from a row pointer already offset
// by (half*8) elements. ISA layout: lane<16 holds M=lane, K in {0..7, 16..23};
// lane>=16 holds M=lane-16, K in {8..15, 24..31}.
__device__ __forceinline__ v16bf load_a_frag(const bf16* p) {
    v8bf lo = *(const v8bf*)p;
    v8bf hi = *(const v8bf*)(p + 16);
    return __builtin_shufflevector(lo, hi, 0,1,2,3,4,5,6,7,8,9,10,11,12,13,14,15);
}

// Stage one 32-byte B-fragment chunk (16 bf16) for this lane into LDS.
__device__ __forceinline__ void stage_b(const bf16* __restrict__ gsrc, bf16* ldst) {
#if HAS_ASYNC_LDS
    __builtin_amdgcn_global_load_async_to_lds_b128(
        (gv4i*)gsrc, (lv4i*)ldst, 0, 0);
    __builtin_amdgcn_global_load_async_to_lds_b128(
        (gv4i*)(gsrc + 8), (lv4i*)(ldst + 8), 0, 0);
#else
    *(v16bf*)ldst = *(const v16bf*)gsrc;
#endif
}

__device__ __forceinline__ void async_stage_wait() {
#if HAS_ASYNC_LDS
    asm volatile("s_wait_asynccnt 0" ::: "memory");
#endif
}

// -------- weight repack: W[K,N] f32 row-major -> B-fragment-ordered bf16 ----
// frag = kt*(N/16)+nt; per frag: 32 lanes x 16 bf16 contiguous.
// lane element i maps to n = nt*16 + (lane&15),
//   k = kt*32 + (lane>=16 ? 8 : 0) + (i<8 ? i : i+8)   (mirror of A layout)
__global__ void pack_w_kernel(const float* __restrict__ W, bf16* __restrict__ out,
                              int K, int N) {
    int idx = blockIdx.x * 256 + threadIdx.x;
    if (idx >= K * N) return;
    int i    = idx & 15;
    int lane = (idx >> 4) & 31;
    int frag = idx >> 9;
    int ntiles = N >> 4;
    int nt = frag % ntiles;
    int kt = frag / ntiles;
    int n = nt * 16 + (lane & 15);
    int k = kt * 32 + ((lane >> 4) << 3) + (i < 8 ? i : i + 8);
    out[idx] = (bf16)W[(size_t)k * N + n];
}

// -------- x init: copy f32, make bf16 shadow --------------------------------
__global__ void x_init_kernel(const float* __restrict__ xin, float* __restrict__ xf,
                              bf16* __restrict__ xb, int n) {
    int i = blockIdx.x * 256 + threadIdx.x;
    if (i >= n) return;
    float v = xin[i];
    xf[i] = v;
    xb[i] = (bf16)v;
}

__global__ void zero_kernel(float* __restrict__ p, int n) {
    int i = blockIdx.x * 256 + threadIdx.x;
    if (i < n) p[i] = 0.0f;
}

// -------- edge embedder: e = GELU(attr@W1+b1)@W2+b2, fused via LDS + WMMA ---
// block = 128 threads (4 waves), 16 edges per block.
__global__ void edge_embed_kernel(const float* __restrict__ attr,
                                  const float* __restrict__ w1,   // [4,128]
                                  const float* __restrict__ b1,   // [128]
                                  const bf16*  __restrict__ w2p,  // packed [128,128]
                                  const float* __restrict__ b2,   // [128]
                                  bf16* __restrict__ eout) {      // [E,128]
    __shared__ bf16 sh[16 * 128];
    const int tid = threadIdx.x;
    const int eb  = blockIdx.x * 16;

    // stage 1: hidden layer into LDS (bf16)
    for (int idx = tid; idx < 16 * 128; idx += 128) {
        int le = idx >> 7;
        int t  = idx & 127;
        const float* a = attr + (size_t)(eb + le) * 4;
        float v = b1[t] + a[0] * w1[t] + a[1] * w1[128 + t]
                        + a[2] * w1[256 + t] + a[3] * w1[384 + t];
        sh[le * 128 + t] = (bf16)gelu_f(v);
    }
    __syncthreads();

    // stage 2: [16,128] x [128,128] WMMA; each wave owns 32 output cols
    const int lane = tid & 31, wave = tid >> 5;
    const int m = lane & 15, half = lane >> 4;
    v8f acc[2] = {};
    const bf16* arow = sh + m * 128 + half * 8;
#pragma unroll
    for (int kt = 0; kt < 4; ++kt) {
        v16bf a = load_a_frag(arow + kt * 32);
#pragma unroll
        for (int nt = 0; nt < 2; ++nt) {
            int frag = kt * 8 + wave * 2 + nt;
            v16bf b = *(const v16bf*)(w2p + (size_t)frag * 512 + lane * 16);
            acc[nt] = wmma_bf16(a, b, acc[nt]);
        }
    }
#pragma unroll
    for (int nt = 0; nt < 2; ++nt) {
        int col = wave * 32 + nt * 16 + m;
        float bb = b2[col];
#pragma unroll
        for (int r = 0; r < 8; ++r) {
            int row = r + half * 8;
            eout[(size_t)(eb + row) * 128 + col] = (bf16)(acc[nt][r] + bb);
        }
    }
}

// -------- dense node GEMM (ff1/ff2): out = [gelu](A@W + b) [+ residual] -----
// WG = 128 threads (4 waves). Tile: 64 rows x 64 cols; waves split M so the
// 4 B-fragments per k-tile are shared through LDS (double-buffered, staged
// with async global->LDS loads). K = 256 (8 k-tiles).
__global__ void ff_gemm_kernel(const bf16* __restrict__ A,
                               const bf16* __restrict__ Wp,
                               const float* __restrict__ bias,
                               const float* __restrict__ residual, // may be null
                               float* __restrict__ outF,
                               bf16*  __restrict__ outB,
                               int apply_gelu) {
    __shared__ bf16 sbuf[2][4 * 512];   // 2 x 4KB: 4 B-frags per k-tile
    const int tid = threadIdx.x;
    const int lane = tid & 31, wave = tid >> 5;
    const int m = lane & 15, half = lane >> 4;
    const int rb  = blockIdx.x * 64 + wave * 16;  // this wave's 16 rows
    const int ntb = blockIdx.y * 4;               // n-tile base (4 tiles = 64 cols)
    const bf16* arow = A + (size_t)(rb + m) * 256 + half * 8;
    const int soff = (wave * 32 + lane) * 16;     // this thread's LDS chunk

    // prologue: stage k-tile 0
    stage_b(Wp + ((size_t)(0 * 16 + ntb + wave) * 512 + lane * 16), &sbuf[0][soff]);
    async_stage_wait();
    __syncthreads();

    v8f acc[4] = {};
#pragma unroll
    for (int kt = 0; kt < 8; ++kt) {
        const int p = kt & 1;
        if (kt + 1 < 8)
            stage_b(Wp + ((size_t)((kt + 1) * 16 + ntb + wave) * 512 + lane * 16),
                    &sbuf[p ^ 1][soff]);
        v16bf a = load_a_frag(arow + kt * 32);
#pragma unroll
        for (int nt = 0; nt < 4; ++nt) {
            v16bf b = *(const v16bf*)&sbuf[p][(nt * 32 + lane) * 16];
            acc[nt] = wmma_bf16(a, b, acc[nt]);
        }
        async_stage_wait();
        __syncthreads();
    }

#pragma unroll
    for (int nt = 0; nt < 4; ++nt) {
        int col = (ntb + nt) * 16 + m;
        float bb = bias[col];
#pragma unroll
        for (int r = 0; r < 8; ++r) {
            int row = rb + r + half * 8;
            size_t o = (size_t)row * 256 + col;
            float v = acc[nt][r] + bb;
            if (apply_gelu) v = gelu_f(v);
            if (residual)   v += residual[o];
            outF[o] = v;
            outB[o] = (bf16)v;
        }
    }
}

// -------- message-pass edge GEMM + scatter-reduce ---------------------------
// m = GELU([h[src] | h[dst] | e] @ W + b); agg[dst] += m  (f32 atomics to L2)
// WG = 128 threads (4 waves). Tile: 64 edges x 64 cols; waves split M, B
// shared through double-buffered LDS. K = 256+256+128 (20 k-tiles), the three
// K-regions read A from h[src], h[dst], e respectively (no concat in memory).
__global__ void mp_edge_kernel(const bf16* __restrict__ hB,   // [N,256]
                               const bf16* __restrict__ eB,   // [E,128]
                               const int*  __restrict__ src,
                               const int*  __restrict__ dst,
                               const bf16* __restrict__ Wp,   // packed [640,256]
                               const float* __restrict__ bias,
                               float* __restrict__ agg) {     // [N,256]
    __shared__ bf16 sbuf[2][4 * 512];
    const int tid = threadIdx.x;
    const int lane = tid & 31, wave = tid >> 5;
    const int m = lane & 15, half = lane >> 4;
    const int eb  = blockIdx.x * 64 + wave * 16;  // this wave's 16 edges
    const int ntb = blockIdx.y * 4;
    const int soff = (wave * 32 + lane) * 16;

    const int sIdx = src[eb + m];
    const int dIdx = dst[eb + m];
    const bf16* srow = hB + (size_t)sIdx * 256 + half * 8;
    const bf16* drow = hB + (size_t)dIdx * 256 + half * 8;
    const bf16* erow = eB + (size_t)(eb + m) * 128 + half * 8;

    stage_b(Wp + ((size_t)(0 * 16 + ntb + wave) * 512 + lane * 16), &sbuf[0][soff]);
    async_stage_wait();
    __syncthreads();

    v8f acc[4] = {};
#pragma unroll
    for (int kt = 0; kt < 20; ++kt) {
        const int p = kt & 1;
        if (kt + 1 < 20)
            stage_b(Wp + ((size_t)((kt + 1) * 16 + ntb + wave) * 512 + lane * 16),
                    &sbuf[p ^ 1][soff]);
        const bf16* ap = (kt < 8)  ? (srow + kt * 32)
                       : (kt < 16) ? (drow + (kt - 8) * 32)
                                   : (erow + (kt - 16) * 32);
        v16bf a = load_a_frag(ap);
#pragma unroll
        for (int nt = 0; nt < 4; ++nt) {
            v16bf b = *(const v16bf*)&sbuf[p][(nt * 32 + lane) * 16];
            acc[nt] = wmma_bf16(a, b, acc[nt]);
        }
        async_stage_wait();
        __syncthreads();
    }

    // epilogue: bias + GELU + scatter-add into agg[dst node]
    int dn[8];
#pragma unroll
    for (int r = 0; r < 8; ++r) dn[r] = dst[eb + r + half * 8];
#pragma unroll
    for (int nt = 0; nt < 4; ++nt) {
        int col = (ntb + nt) * 16 + m;
        float bb = bias[col];
#pragma unroll
        for (int r = 0; r < 8; ++r) {
            float v = gelu_f(acc[nt][r] + bb);
            atomicAdd(agg + (size_t)dn[r] * 256 + col, v);
        }
    }
}

// -------- h += agg; refresh bf16 shadow; re-zero agg ------------------------
__global__ void h_update_kernel(float* __restrict__ h, bf16* __restrict__ hB,
                                float* __restrict__ agg, int n) {
    int i = blockIdx.x * 256 + threadIdx.x;
    if (i >= n) return;
    float v = h[i] + agg[i];
    h[i] = v;
    hB[i] = (bf16)v;
    agg[i] = 0.0f;
}

extern "C" void kernel_launch(void* const* d_in, const int* in_sizes, int n_in,
                              void* d_out, int out_size, void* d_ws, size_t ws_size,
                              hipStream_t stream) {
    const float* x_in   = (const float*)d_in[0];
    const int*   eidx   = (const int*)d_in[1];
    const float* eattr  = (const float*)d_in[2];
    const float* ee_w1  = (const float*)d_in[3];
    const float* ee_b1  = (const float*)d_in[4];
    const float* ee_w2  = (const float*)d_in[5];
    const float* ee_b2  = (const float*)d_in[6];
    const float* ff1_w  = (const float*)d_in[7];
    const float* ff1_b  = (const float*)d_in[8];
    const float* mp1_w  = (const float*)d_in[9];
    const float* mp1_b  = (const float*)d_in[10];
    const float* mp2_w  = (const float*)d_in[11];
    const float* mp2_b  = (const float*)d_in[12];
    const float* ff2_w  = (const float*)d_in[13];
    const float* ff2_b  = (const float*)d_in[14];
    const int* src = eidx;
    const int* dst = eidx + NE;

    char* ws = (char*)d_ws;
    auto alloc = [&](size_t bytes) {
        char* p = ws;
        ws += (bytes + 255) & ~(size_t)255;
        return p;
    };
    bf16*  x_bf  = (bf16*) alloc((size_t)NN * 256 * 2);
    float* h_f   = (float*)alloc((size_t)NN * 256 * 4);
    bf16*  h_bf  = (bf16*) alloc((size_t)NN * 256 * 2);
    float* agg   = (float*)alloc((size_t)NN * 256 * 4);
    bf16*  e_bf  = (bf16*) alloc((size_t)NE * 128 * 2);
    bf16*  pk_ee = (bf16*) alloc((size_t)128 * 128 * 2);
    bf16*  pk_f1 = (bf16*) alloc((size_t)4 * 256 * 256 * 2);
    bf16*  pk_f2 = (bf16*) alloc((size_t)4 * 256 * 256 * 2);
    bf16*  pk_m1 = (bf16*) alloc((size_t)4 * 640 * 256 * 2);
    bf16*  pk_m2 = (bf16*) alloc((size_t)4 * 640 * 256 * 2);
    float* x_f   = (float*)d_out;

    const int NELEM = NN * 256;

    // repack all weights into B-fragment order (bf16)
    pack_w_kernel<<<(128 * 128 + 255) / 256, 256, 0, stream>>>(ee_w2, pk_ee, 128, 128);
    for (int d = 0; d < 4; ++d) {
        pack_w_kernel<<<(256 * 256 + 255) / 256, 256, 0, stream>>>(
            ff1_w + (size_t)d * 256 * 256, pk_f1 + (size_t)d * 256 * 256, 256, 256);
        pack_w_kernel<<<(256 * 256 + 255) / 256, 256, 0, stream>>>(
            ff2_w + (size_t)d * 256 * 256, pk_f2 + (size_t)d * 256 * 256, 256, 256);
        pack_w_kernel<<<(640 * 256 + 255) / 256, 256, 0, stream>>>(
            mp1_w + (size_t)d * 640 * 256, pk_m1 + (size_t)d * 640 * 256, 640, 256);
        pack_w_kernel<<<(640 * 256 + 255) / 256, 256, 0, stream>>>(
            mp2_w + (size_t)d * 640 * 256, pk_m2 + (size_t)d * 640 * 256, 640, 256);
    }

    x_init_kernel<<<(NELEM + 255) / 256, 256, 0, stream>>>(x_in, x_f, x_bf, NELEM);
    zero_kernel<<<(NELEM + 255) / 256, 256, 0, stream>>>(agg, NELEM);

    edge_embed_kernel<<<NE / 16, 128, 0, stream>>>(eattr, ee_w1, ee_b1, pk_ee, ee_b2, e_bf);

    dim3 ffgrid(NN / 64, 4);
    dim3 mpgrid(NE / 64, 4);
    for (int d = 0; d < 4; ++d) {
        // h = GELU(x @ ff1_w + b)
        ff_gemm_kernel<<<ffgrid, 128, 0, stream>>>(
            x_bf, pk_f1 + (size_t)d * 256 * 256, ff1_b + d * 256,
            nullptr, h_f, h_bf, 1);
        // message pass 1
        mp_edge_kernel<<<mpgrid, 128, 0, stream>>>(
            h_bf, e_bf, src, dst, pk_m1 + (size_t)d * 640 * 256, mp1_b + d * 256, agg);
        h_update_kernel<<<(NELEM + 255) / 256, 256, 0, stream>>>(h_f, h_bf, agg, NELEM);
        // message pass 2
        mp_edge_kernel<<<mpgrid, 128, 0, stream>>>(
            h_bf, e_bf, src, dst, pk_m2 + (size_t)d * 640 * 256, mp2_b + d * 256, agg);
        h_update_kernel<<<(NELEM + 255) / 256, 256, 0, stream>>>(h_f, h_bf, agg, NELEM);
        // x = residual + h @ ff2_w + b   (residual read & write in d_out)
        ff_gemm_kernel<<<ffgrid, 128, 0, stream>>>(
            h_bf, pk_f2 + (size_t)d * 256 * 256, ff2_b + d * 256,
            x_f, x_f, x_bf, 0);
    }
}
